// TieredMemory_75617194213657
// MI455X (gfx1250) — matrix operand order repaired and checked by reading, not compiled, not run on hardware
//
#include <hip/hip_runtime.h>
#include <hip/hip_bf16.h>

// ---------------- problem constants ----------------
#define N_NODES 200000
#define DIM     256
#define WDIM    128
#define BQ      64
#define TOPK    16
#define NTILES  (N_NODES / 16)          // 12500
#define K1_WAVES 4
#define K1_GRID  (NTILES / K1_WAVES)    // 3125
#define CHUNK    256
#define NBLK2    ((N_NODES + CHUNK - 1) / CHUNK)   // 782
#define NEG_MASK (-1.0e30f)
#define FLT_LOW  (-3.0e38f)

// ---------------- d_out layout (floats) ----------------
#define KL_OFF  (51200000L)                      // N*D
#define RET_OFF (KL_OFF + 1L)
#define IDX_OFF (RET_OFF + 64L * 16L * 256L)

// ---------------- ws layout (bytes) ----------------
#define OFF_KL    0L
#define OFF_CNT   8L
#define OFF_INV   64L                            // N floats (800000 B)
#define OFF_WMU   800128L                        // 128*256 bf16 (65536 B)
#define OFF_WLV   865664L
#define OFF_WDEC  931200L
#define OFF_QN    996736L                        // 64*256 bf16 (32768 B)
#define OFF_BTV   1029504L                       // NBLK2*64*16 floats
#define OFF_BTI   4232576L                       // NBLK2*64*16 ints

typedef __attribute__((ext_vector_type(16))) __bf16       v16bf;
typedef __attribute__((ext_vector_type(8)))  float        v8f;
typedef __attribute__((ext_vector_type(8)))  unsigned int v8u;

// ---- bf16 pack (round-to-nearest-even) ----
__device__ __forceinline__ unsigned short f2bf(float f) {
    unsigned u = __float_as_uint(f);
    unsigned r = 0x7FFFu + ((u >> 16) & 1u);
    return (unsigned short)((u + r) >> 16);
}
__device__ __forceinline__ unsigned packbf(float lo, float hi) {
    return (unsigned)f2bf(lo) | ((unsigned)f2bf(hi) << 16);
}

// A-fragment K offset within a 32-wide K block (contiguous: k = 16*half + 2*v, matches B)
__device__ __forceinline__ int a_koff(int v, int half) {
    return 16 * half + 2 * v;
}

__device__ __forceinline__ v8f wmma_bf16(v8u a, v8u b, v8f c) {
    return __builtin_amdgcn_wmma_f32_16x16x32_bf16(
        false, __builtin_bit_cast(v16bf, a),
        false, __builtin_bit_cast(v16bf, b),
        (short)0, c, false, false);
}

// =====================================================================
// Kernel 0: query normalization -> bf16, weight f32 -> bf16, zero accums
// grid = 64 (query rows) + 128 (weight conversion) = 192 blocks x 256
// =====================================================================
__global__ void __launch_bounds__(256)
k_prep(const float* __restrict__ query,
       const float* __restrict__ Wmu, const float* __restrict__ Wlv,
       const float* __restrict__ Wdec,
       unsigned short* __restrict__ qn,
       unsigned short* __restrict__ wmu, unsigned short* __restrict__ wlv,
       unsigned short* __restrict__ wdec,
       float* kl_sum, int* warm_cnt) {
    int tid = threadIdx.x;
    if (blockIdx.x < BQ) {
        __shared__ float red[256];
        int b = blockIdx.x;
        float v = query[b * DIM + tid];
        red[tid] = v * v;
        __syncthreads();
        for (int s = 128; s; s >>= 1) {
            if (tid < s) red[tid] += red[tid + s];
            __syncthreads();
        }
        float inv = 1.0f / (sqrtf(red[0]) + 1e-10f);
        qn[b * DIM + tid] = f2bf(v * inv);
    } else {
        int i = (blockIdx.x - BQ) * 256 + tid;   // [0, 32768)
        wmu[i]  = f2bf(Wmu[i]);
        wlv[i]  = f2bf(Wlv[i]);
        wdec[i] = f2bf(Wdec[i]);
        if (blockIdx.x == BQ && tid == 0) { *kl_sum = 0.0f; *warm_cnt = 0; }
    }
}

// =====================================================================
// Kernel 1: fused mu/logvar/dec/new_features/KL/inv-norm.
// 1 wave = one 16-row node tile. block = 128 (4 waves), grid = 3125.
// =====================================================================
__global__ void __launch_bounds__(128)
k_compress(const float* __restrict__ x, const int* __restrict__ tiers,
           const float* __restrict__ bmu, const float* __restrict__ blv,
           const float* __restrict__ bdec,
           const unsigned short* __restrict__ wmu,
           const unsigned short* __restrict__ wlv,
           const unsigned short* __restrict__ wdec,
           float* __restrict__ newf, float* __restrict__ invn,
           float* kl_sum, int* warm_cnt) {
    __shared__ float mu_lds[K1_WAVES][16][WDIM];   // 32 KB
    __shared__ int   tier_s[K1_WAVES][16];

    const int wave = threadIdx.x >> 5;
    const int lane = threadIdx.x & 31;
    const int half = lane >> 4;
    const int l16  = lane & 15;
    const int tile = blockIdx.x * K1_WAVES + wave;   // exact: 12500 tiles
    const long base = (long)tile * 16;

    if (lane < 16) tier_s[wave][lane] = tiers[base + lane];

    // ---- A fragments: node_features rows -> bf16 (8 K-blocks of 32) ----
    v8u a_raw[8];
    const float* xrow = x + (base + l16) * DIM;
    for (int kb = 0; kb < 8; ++kb)
        for (int v = 0; v < 8; ++v) {
            int k = kb * 32 + a_koff(v, half);
            a_raw[kb][v] = packbf(xrow[k], xrow[k + 1]);
        }

    // ---- mu & logvar tiles (N = 128 -> 8 n-tiles), KL, stage mu in LDS ----
    float kl_local = 0.0f;
    for (int nt = 0; nt < 8; ++nt) {
        v8f mu_acc = {};
        v8f lv_acc = {};
        const int ncol = nt * 16 + l16;            // mu/lv column
        const unsigned short* wmrow = wmu + (long)ncol * DIM;
        const unsigned short* wlrow = wlv + (long)ncol * DIM;
        for (int kb = 0; kb < 8; ++kb) {
            v8u bm, bl;
            for (int v = 0; v < 8; ++v) {
                int k = kb * 32 + 16 * half + 2 * v;
                bm[v] = *(const unsigned*)(wmrow + k);
                bl[v] = *(const unsigned*)(wlrow + k);
            }
            mu_acc = wmma_bf16(a_raw[kb], bm, mu_acc);
            lv_acc = wmma_bf16(a_raw[kb], bl, lv_acc);
        }
        const float bmv = bmu[ncol], blv_ = blv[ncol];
        for (int r = 0; r < 8; ++r) {
            int m = r + 8 * half;
            float muv = mu_acc[r] + bmv;
            float lvv = lv_acc[r] + blv_;
            mu_lds[wave][m][ncol] = muv;
            if (tier_s[wave][m] == 1)
                kl_local += -0.5f * (1.0f + lvv - muv * muv - __expf(lvv));
        }
    }
    for (int off = 16; off; off >>= 1) kl_local += __shfl_xor(kl_local, off, 32);
    if (lane == 0) {
        atomicAdd(kl_sum, kl_local);
        int c = 0;
        for (int m = 0; m < 16; ++m) c += (tier_s[wave][m] == 1);
        atomicAdd(warm_cnt, c);
    }

    // ---- dec = mu @ W_dec.T  (K = 128 -> 4 K-blocks), select, store ----
    v8u a2[4];
    for (int kb = 0; kb < 4; ++kb)
        for (int v = 0; v < 8; ++v) {
            int k = kb * 32 + a_koff(v, half);
            a2[kb][v] = packbf(mu_lds[wave][l16][k], mu_lds[wave][l16][k + 1]);
        }

    bool warm_r[8];
    for (int r = 0; r < 8; ++r) warm_r[r] = (tier_s[wave][r + 8 * half] == 1);

    float sq[8] = {0, 0, 0, 0, 0, 0, 0, 0};
    for (int nt = 0; nt < 16; ++nt) {
        v8f acc = {};
        const int col = nt * 16 + l16;             // output column (D)
        const unsigned short* wdrow = wdec + (long)col * WDIM;
        for (int kb = 0; kb < 4; ++kb) {
            v8u bd;
            for (int v = 0; v < 8; ++v) {
                int k = kb * 32 + 16 * half + 2 * v;
                bd[v] = *(const unsigned*)(wdrow + k);
            }
            acc = wmma_bf16(a2[kb], bd, acc);
        }
        const float bdv = bdec[col];
        for (int r = 0; r < 8; ++r) {
            long row = base + r + 8 * half;
            float orig = x[row * DIM + col];
            float val = warm_r[r] ? (acc[r] + bdv) : orig;
            newf[row * DIM + col] = val;
            sq[r] += val * val;
        }
    }
    for (int r = 0; r < 8; ++r)
        for (int off = 1; off < 16; off <<= 1) sq[r] += __shfl_xor(sq[r], off, 32);
    if (l16 == 0)
        for (int r = 0; r < 8; ++r)
            invn[base + r + 8 * half] = 1.0f / (sqrtf(sq[r]) + 1e-10f);
}

// =====================================================================
// Kernel 2: sim = qn @ fn^T over a 256-node chunk (WMMA), per-chunk top-16
// block = 256 (8 waves), grid = 782. LDS sim tile = 64 KB.
// =====================================================================
__global__ void __launch_bounds__(256)
k_sim(const float* __restrict__ newf, const int* __restrict__ tiers,
      const float* __restrict__ invn, const unsigned short* __restrict__ qn,
      float* __restrict__ btv, int* __restrict__ bti) {
    __shared__ float sim[BQ][CHUNK];               // 65536 B

    const int tid  = threadIdx.x;
    const int lane = tid & 31;
    const int wave = tid >> 5;
    const int half = lane >> 4;
    const int l16  = lane & 15;
    const long cbase = (long)blockIdx.x * CHUNK;

    // A fragments: this wave's query tile (mt = wave & 3)
    const int mt = wave & 3;
    v8u a_raw[8];
    const unsigned short* qrow = qn + (long)(mt * 16 + l16) * DIM;
    for (int kb = 0; kb < 8; ++kb)
        for (int v = 0; v < 8; ++v) {
            int k = kb * 32 + a_koff(v, half);
            a_raw[kb][v] = *(const unsigned*)(qrow + k);
        }

    // node tiles: nt = (wave>>2) + 2*j  covers 0..15 across the 2 wave groups
    for (int j = 0; j < 8; ++j) {
        const int nt = (wave >> 2) + 2 * j;
        const long node = cbase + nt * 16 + l16;
        const bool okn = node < N_NODES;
        const float inv = okn ? invn[node] : 0.0f;
        const float* frow = newf + node * DIM;
        v8f acc = {};
        for (int kb = 0; kb < 8; ++kb) {
            v8u bfrag;
            for (int v = 0; v < 8; ++v) {
                int k = kb * 32 + 16 * half + 2 * v;
                float f0 = okn ? frow[k] * inv : 0.0f;
                float f1 = okn ? frow[k + 1] * inv : 0.0f;
                bfrag[v] = packbf(f0, f1);
            }
            acc = wmma_bf16(a_raw[kb], bfrag, acc);
        }
        for (int r = 0; r < 8; ++r)
            sim[mt * 16 + r + 8 * half][nt * 16 + l16] = acc[r];
    }
    __syncthreads();

    // per-thread top-16 over 64 nodes (4 threads per query)
    const int q = tid >> 2, part = tid & 3;
    float tv[TOPK]; int ti[TOPK];
    for (int j = 0; j < TOPK; ++j) { tv[j] = FLT_LOW; ti[j] = -1; }
    for (int s = 0; s < 64; ++s) {
        const int nl = part * 64 + s;
        const long node = cbase + nl;
        float v = FLT_LOW;
        if (node < N_NODES)
            v = (tiers[node] == 2) ? sim[q][nl] : NEG_MASK;
        if (v > tv[TOPK - 1]) {
            int p = TOPK - 1;
            while (p > 0 && tv[p - 1] < v) { tv[p] = tv[p - 1]; ti[p] = ti[p - 1]; --p; }
            tv[p] = v; ti[p] = (int)node;
        }
    }
    __syncthreads();                                 // sim reads done

    // overlay candidate lists into the sim LDS (64q x 64 cands)
    float* cv = &sim[0][0];                          // 4096 floats
    int*   ci = (int*)(&sim[0][0] + 4096);           // 4096 ints
    for (int j = 0; j < TOPK; ++j) {
        cv[q * 64 + part * TOPK + j] = tv[j];
        ci[q * 64 + part * TOPK + j] = ti[j];
    }
    __syncthreads();

    if (tid < BQ) {
        float* v4 = cv + tid * 64;
        int*   i4 = ci + tid * 64;
        const long slot = ((long)blockIdx.x * BQ + tid) * TOPK;
        for (int j = 0; j < TOPK; ++j) {
            int best = 0; float bv = v4[0];
            for (int s = 1; s < 64; ++s)
                if (v4[s] > bv) { bv = v4[s]; best = s; }
            btv[slot + j] = bv;
            bti[slot + j] = i4[best];
            v4[best] = FLT_LOW;
        }
    }
}

// =====================================================================
// Kernel 3: per-query merge of 782*16 candidates, KL finalize, gather.
// block = 256 (one query per block), grid = 64.
// =====================================================================
__global__ void __launch_bounds__(256)
k_final(const float* __restrict__ btv, const int* __restrict__ bti,
        const float* __restrict__ kl_sum, const int* __restrict__ warm_cnt,
        float* __restrict__ out) {
    const int b = blockIdx.x, tid = threadIdx.x;
    if (b == 0 && tid == 0) {
        int c = *warm_cnt; if (c < 1) c = 1;
        out[KL_OFF] = *kl_sum / ((float)c * (float)WDIM);
    }

    float tv[TOPK]; int ti[TOPK];
    for (int j = 0; j < TOPK; ++j) { tv[j] = FLT_LOW; ti[j] = -1; }
    const int CAND = NBLK2 * TOPK;                   // 12512
    for (int i = tid; i < CAND; i += 256) {
        const int blk = i >> 4, j = i & 15;
        const long o = ((long)blk * BQ + b) * TOPK + j;
        float v = btv[o];
        if (v > tv[TOPK - 1]) {
            int ix = bti[o];
            int p = TOPK - 1;
            while (p > 0 && tv[p - 1] < v) { tv[p] = tv[p - 1]; ti[p] = ti[p - 1]; --p; }
            tv[p] = v; ti[p] = ix;
        }
    }

    __shared__ float cv[256 * TOPK];                 // 16 KB
    __shared__ int   ci[256 * TOPK];                 // 16 KB
    __shared__ int   fin[TOPK];
    for (int j = 0; j < TOPK; ++j) { cv[tid * TOPK + j] = tv[j]; ci[tid * TOPK + j] = ti[j]; }
    __syncthreads();

    if (tid == 0) {
        for (int j = 0; j < TOPK; ++j) {
            int best = 0; float bv = cv[0];
            for (int s = 1; s < 256 * TOPK; ++s)
                if (cv[s] > bv) { bv = cv[s]; best = s; }
            int ix = ci[best];
            if (ix < 0) ix = 0;
            fin[j] = ix;
            out[IDX_OFF + (long)b * TOPK + j] = (float)ix;
            cv[best] = FLT_LOW;
        }
    }
    __syncthreads();

    for (int j = 0; j < TOPK; ++j) {
        const long row = fin[j];
        out[RET_OFF + ((long)b * TOPK + j) * DIM + tid] = out[row * DIM + tid];
    }
}

// =====================================================================
extern "C" void kernel_launch(void* const* d_in, const int* in_sizes, int n_in,
                              void* d_out, int out_size, void* d_ws, size_t ws_size,
                              hipStream_t stream) {
    (void)in_sizes; (void)n_in; (void)out_size; (void)ws_size;

    const float* x     = (const float*)d_in[0];
    const int*   tiers = (const int*)  d_in[1];
    const float* query = (const float*)d_in[2];
    const float* Wmu   = (const float*)d_in[3];
    const float* bmu   = (const float*)d_in[4];
    const float* Wlv   = (const float*)d_in[5];
    const float* blv   = (const float*)d_in[6];
    const float* Wdec  = (const float*)d_in[7];
    const float* bdec  = (const float*)d_in[8];
    // d_in[9] = top_k, fixed at 16 for this problem.

    char* ws = (char*)d_ws;
    float*          kl_sum   = (float*)(ws + OFF_KL);
    int*            warm_cnt = (int*)  (ws + OFF_CNT);
    float*          invn     = (float*)(ws + OFF_INV);
    unsigned short* wmu_bf   = (unsigned short*)(ws + OFF_WMU);
    unsigned short* wlv_bf   = (unsigned short*)(ws + OFF_WLV);
    unsigned short* wdec_bf  = (unsigned short*)(ws + OFF_WDEC);
    unsigned short* qn_bf    = (unsigned short*)(ws + OFF_QN);
    float*          btv      = (float*)(ws + OFF_BTV);
    int*            bti      = (int*)  (ws + OFF_BTI);

    float* out = (float*)d_out;

    k_prep<<<192, 256, 0, stream>>>(query, Wmu, Wlv, Wdec,
                                    qn_bf, wmu_bf, wlv_bf, wdec_bf,
                                    kl_sum, warm_cnt);

    k_compress<<<K1_GRID, 32 * K1_WAVES, 0, stream>>>(
        x, tiers, bmu, blv, bdec, wmu_bf, wlv_bf, wdec_bf,
        out /* new_features at offset 0 */, invn, kl_sum, warm_cnt);

    k_sim<<<NBLK2, 256, 0, stream>>>(out, tiers, invn, qn_bf, btv, bti);

    k_final<<<BQ, 256, 0, stream>>>(btv, bti, kl_sum, warm_cnt, out);
}
